// RSSM_63118839382683
// MI455X (gfx1250) — matrix operand order repaired and checked by reading, not compiled, not run on hardware
//
#include <hip/hip_runtime.h>
#include <hip/hip_bf16.h>
#include <cstdint>

// ---------- types ----------
typedef __bf16 bf16_t;
typedef __attribute__((ext_vector_type(16))) __bf16 v16bf;
typedef __attribute__((ext_vector_type(8)))  float  v8f;
typedef __attribute__((ext_vector_type(4)))  unsigned int u32x4;

union FragBF { u32x4 q[2]; v16bf v; };

// ---------- constants ----------
#define TT 64
#define BB 256
#define EE 1024
#define AA 6
#define HH 1024
#define KPRE_PAD 1056   // 1030 padded to mult of 32

#define GTM 128
#define GTN 128
#define GTK 32
#define APAD 8
#define BPAD 8

// =====================================================================
// WMMA GEMM:  C[256,N] = act( A_f32[256,K] @ Wt_bf16^T + bias )
//   Wt is stored transposed: Wt[n*K + k], K multiple of 32.
//   Block tile 128x128, 8 waves (wave32), each wave 4x2 tiles of 16x16.
//   Software-pipelined: weights via GLOBAL_LOAD_ASYNC_TO_LDS_B128 (ASYNCcnt),
//   activations prefetched into VGPRs a chunk ahead, double-buffered LDS.
// =====================================================================
__global__ void __launch_bounds__(256)
gemm_bf16_wmma(const float* __restrict__ A, int lda,
               const bf16_t* __restrict__ Wt, int K,
               const float* __restrict__ bias,
               float* __restrict__ C, int ldc, int act)
{
    __shared__ bf16_t As[2][GTM][GTK + APAD];
    __shared__ bf16_t Bs[2][GTN][GTK + BPAD];

    const int tid   = threadIdx.x;
    const int lane  = tid & 31;
    const int w     = tid >> 5;      // 0..7
    const int waveM = w >> 2;        // 0..1
    const int waveN = w & 3;         // 0..3
    const int waveMrow = waveM * 64; // LDS-local
    const int waveNcol = waveN * 32; // LDS-local
    const int rowBase  = blockIdx.y * GTM + waveMrow;
    const int colBase  = blockIdx.x * GTN + waveNcol;

    // A-fill mapping: thread owns rows {p*32 + tid>>3}, cols [(tid&7)*4 .. +3]
    const int aRow = tid >> 3;
    const int aCol = (tid & 7) * 4;
    const float* Abase = A + (size_t)(blockIdx.y * GTM) * lda;

    // B-fill mapping: thread issues 2 async b128: n = p*64 + tid>>2, k = (tid&3)*8
    const int bN = tid >> 2;
    const int bK = (tid & 3) * 8;
    const bf16_t* Wbase = Wt + (size_t)(blockIdx.x * GTN) * K;

    v8f acc[4][2];
#pragma unroll
    for (int i = 0; i < 4; ++i)
#pragma unroll
        for (int j = 0; j < 2; ++j)
#pragma unroll
            for (int r = 0; r < 8; ++r) acc[i][j][r] = 0.0f;

    const int halfsel = lane >> 4;   // 0 or 1
    const int l15     = lane & 15;

    // ---- pipeline helpers ----
    auto issueB = [&](int buf, int k0) {
#pragma unroll
        for (int p = 0; p < 2; ++p) {
            int n = p * 64 + bN;
            unsigned ldsb = (unsigned)(uintptr_t)&Bs[buf][n][bK];
            unsigned long long ga =
                (unsigned long long)(uintptr_t)(Wbase + (size_t)n * K + k0 + bK);
            asm volatile("global_load_async_to_lds_b128 %0, %1, off"
                         :: "v"(ldsb), "v"(ga) : "memory");
        }
    };
    float4 ar[4];
    auto loadA = [&](int k0) {
#pragma unroll
        for (int p = 0; p < 4; ++p)
            ar[p] = *reinterpret_cast<const float4*>(
                Abase + (size_t)(p * 32 + aRow) * lda + k0 + aCol);
    };
    auto storeA = [&](int buf) {
#pragma unroll
        for (int p = 0; p < 4; ++p) {
            int r = p * 32 + aRow;
            As[buf][r][aCol + 0] = (bf16_t)ar[p].x;
            As[buf][r][aCol + 1] = (bf16_t)ar[p].y;
            As[buf][r][aCol + 2] = (bf16_t)ar[p].z;
            As[buf][r][aCol + 3] = (bf16_t)ar[p].w;
        }
    };

    // ---- prologue: stage chunk 0 ----
    issueB(0, 0);
    loadA(0);

    const int nIter = K / GTK;
    for (int it = 0; it < nIter; ++it) {
        const int buf = it & 1;

        // commit staged A chunk to LDS; ensure async B chunk has landed
        storeA(buf);
        asm volatile("s_wait_asynccnt 0x0" ::: "memory");
        __syncthreads();

        // stage next chunk while this one computes
        if (it + 1 < nIter) {
            issueB(buf ^ 1, (it + 1) * GTK);
            loadA((it + 1) * GTK);
        }
        if (it + 2 < nIter) {
            __builtin_prefetch(
                Wbase + (size_t)(tid >> 1) * K + (it + 2) * GTK, 0, 0);
        }

        // ---- fragment loads from LDS (CDNA5 16-bit A/B striping) ----
        FragBF afr[4], bfr[2];
#pragma unroll
        for (int tm = 0; tm < 4; ++tm) {
            int r  = waveMrow + tm * 16 + l15;
            int kb = halfsel * 8;               // lanes<16: K0..7 / lanes>=16: K8..15
            afr[tm].q[0] = *(const u32x4*)&As[buf][r][kb];
            afr[tm].q[1] = *(const u32x4*)&As[buf][r][kb + 16]; // K16..23 / K24..31
        }
#pragma unroll
        for (int tn = 0; tn < 2; ++tn) {
            int n  = waveNcol + tn * 16 + l15;
            int kb = halfsel * 8;               // lanes<16: K0..7 / lanes>=16: K8..15
            bfr[tn].q[0] = *(const u32x4*)&Bs[buf][n][kb];
            bfr[tn].q[1] = *(const u32x4*)&Bs[buf][n][kb + 16]; // K16..23 / K24..31
        }
#pragma unroll
        for (int tm = 0; tm < 4; ++tm)
#pragma unroll
            for (int tn = 0; tn < 2; ++tn)
                acc[tm][tn] = __builtin_amdgcn_wmma_f32_16x16x32_bf16(
                    false, afr[tm].v, false, bfr[tn].v,
                    (short)0, acc[tm][tn], false, false);
    }

    // ---- epilogue: bias + optional ELU, scatter per C/D layout ----
    const int rAdd = halfsel * 8;   // lanes>=16 hold M = r+8
#pragma unroll
    for (int tm = 0; tm < 4; ++tm) {
#pragma unroll
        for (int tn = 0; tn < 2; ++tn) {
            int col = colBase + tn * 16 + l15;
            float bv = bias[col];
#pragma unroll
            for (int r = 0; r < 8; ++r) {
                int row = rowBase + tm * 16 + rAdd + r;
                float v = acc[tm][tn][r] + bv;
                if (act) v = (v > 0.0f) ? v : (__expf(v) - 1.0f);
                C[(size_t)row * ldc + col] = v;
            }
        }
    }
}

// =====================================================================
// Weight transpose + fp32->bf16 convert:  Wt[n*Kpad + k] = W[k*N + n]
// =====================================================================
__global__ void convert_wt(const float* __restrict__ W, bf16_t* __restrict__ Wt,
                           int K, int N, int Kpad)
{
    long i = (long)blockIdx.x * blockDim.x + threadIdx.x;
    long total = (long)N * Kpad;
    if (i >= total) return;
    int n = (int)(i / Kpad), k = (int)(i % Kpad);
    Wt[i] = (k < K) ? (bf16_t)W[(long)k * N + n] : (bf16_t)0.0f;
}

__global__ void zero_f32(float* __restrict__ p, long n)
{
    long i = (long)blockIdx.x * blockDim.x + threadIdx.x;
    if (i < n) p[i] = 0.0f;
}

// x_in[b, 0:1024]=stoch, [1024:1030]=action_t, [1030:1056]=0
__global__ void concat_xin(const float* __restrict__ state,
                           const float* __restrict__ act_t,
                           float* __restrict__ xin)
{
    int i = blockIdx.x * blockDim.x + threadIdx.x;
    if (i >= BB * KPRE_PAD) return;
    int b = i / KPRE_PAD, j = i % KPRE_PAD;
    float v;
    if (j < 1024)       v = state[(size_t)b * 2048 + 1024 + j];  // stoch half
    else if (j < 1030)  v = act_t[(size_t)b * AA + (j - 1024)];
    else                v = 0.0f;
    xin[i] = v;
}

// out[b,0:1024] = a[b*lda + aoff + j]; out[b,1024:2048] = bsrc[b*ldb + j-1024]
__global__ void concat2(const float* __restrict__ a, int lda_, int aoff,
                        const float* __restrict__ bsrc, int ldb_,
                        float* __restrict__ out)
{
    int i = blockIdx.x * blockDim.x + threadIdx.x;
    if (i >= BB * 2048) return;
    int b = i >> 11, j = i & 2047;
    out[i] = (j < 1024) ? a[(size_t)b * lda_ + aoff + j]
                        : bsrc[(size_t)b * ldb_ + (j - 1024)];
}

// =====================================================================
// LayerNorm(3072) + GRU gates, writes new deter to state[:, :1024] and
// to the feature output slice. One block per batch row.
// =====================================================================
__global__ void __launch_bounds__(256)
gru_ln_kernel(const float* __restrict__ parts,
              const float* __restrict__ ln_scale,
              const float* __restrict__ ln_offset,
              float* __restrict__ state,   // [256,2048]
              float* __restrict__ feat)    // [256,2048] (slice t)
{
    int b = blockIdx.x, tid = threadIdx.x;
    const float* p = parts + (size_t)b * 3072;

    float s = 0.0f, ss = 0.0f;
    for (int j = tid; j < 3072; j += 256) {
        float v = p[j]; s += v; ss += v * v;
    }
    for (int off = 16; off; off >>= 1) {
        s  += __shfl_xor(s,  off);
        ss += __shfl_xor(ss, off);
    }
    __shared__ float red0[8], red1[8];
    if ((tid & 31) == 0) { red0[tid >> 5] = s; red1[tid >> 5] = ss; }
    __syncthreads();
    s = 0.0f; ss = 0.0f;
#pragma unroll
    for (int i = 0; i < 8; ++i) { s += red0[i]; ss += red1[i]; }

    const float mu  = s * (1.0f / 3072.0f);
    const float var = ss * (1.0f / 3072.0f) - mu * mu;
    const float inv = rsqrtf(var + 1e-5f);

    float* deter = state + (size_t)b * 2048;
    float* f     = feat  + (size_t)b * 2048;
    for (int j = tid; j < 1024; j += 256) {
        float l0 = (p[j]        - mu) * inv * ln_scale[j]        + ln_offset[j];
        float l1 = (p[1024 + j] - mu) * inv * ln_scale[1024 + j] + ln_offset[1024 + j];
        float l2 = (p[2048 + j] - mu) * inv * ln_scale[2048 + j] + ln_offset[2048 + j];
        float reset = 1.0f / (1.0f + __expf(-l0));
        float cand  = tanhf(reset * l1);
        float upd   = 1.0f / (1.0f + __expf(-(l2 - 1.0f)));
        float nd    = upd * cand + (1.0f - upd) * deter[j];
        deter[j] = nd;
        f[j]     = nd;
    }
}

// =====================================================================
// Straight-through categorical sample == one-hot numerically.
// Gumbel-max per 32-way group; wave32 reduction (lane == D index).
// =====================================================================
__device__ __forceinline__ unsigned hash_u32(unsigned x)
{
    x ^= x >> 16; x *= 0x7feb352dU;
    x ^= x >> 15; x *= 0x846ca68bU;
    x ^= x >> 16; return x;
}

__global__ void __launch_bounds__(256)
sample_kernel(const float* __restrict__ stats,  // [256,1024] slice
              float* __restrict__ state,        // [256,2048]
              float* __restrict__ feat,         // [256,2048] slice t
              unsigned seed)
{
    int b = blockIdx.x, tid = threadIdx.x;
    int d = tid & 31;
#pragma unroll
    for (int p = 0; p < 4; ++p) {
        int sgrp = p * 8 + (tid >> 5);
        float logit = stats[(size_t)b * 1024 + sgrp * 32 + d];
        unsigned h = hash_u32(seed ^ (unsigned)(b * 0x0001F7F7u)
                                   ^ (unsigned)(sgrp * 0x9E3779B9u)
                                   ^ (unsigned)(d * 0x85EBCA6Bu));
        float u = (float)(h >> 8) * (1.0f / 16777216.0f) + 1e-7f;
        float g = -__logf(-__logf(u));
        float key = logit + g;
        int idx = d;
        for (int off = 16; off; off >>= 1) {
            float ok = __shfl_xor(key, off);
            int   oi = __shfl_xor(idx, off);
            if (ok > key || (ok == key && oi < idx)) { key = ok; idx = oi; }
        }
        float v = (idx == d) ? 1.0f : 0.0f;
        state[(size_t)b * 2048 + 1024 + sgrp * 32 + d] = v;
        feat [(size_t)b * 2048 + 1024 + sgrp * 32 + d] = v;
    }
}

// =====================================================================
// host orchestration
// =====================================================================
extern "C" void kernel_launch(void* const* d_in, const int* in_sizes, int n_in,
                              void* d_out, int out_size, void* d_ws, size_t ws_size,
                              hipStream_t stream)
{
    (void)in_sizes; (void)n_in; (void)out_size; (void)ws_size;

    const float* embeds    = (const float*)d_in[0];   // [64,256,1024]
    const float* actions   = (const float*)d_in[1];   // [64,256,6]
    const float* W_pre     = (const float*)d_in[2];   // [1030,1024]
    const float* b_pre     = (const float*)d_in[3];
    const float* W_gru     = (const float*)d_in[4];   // [2048,3072]
    const float* b_gru     = (const float*)d_in[5];
    const float* ln_scale  = (const float*)d_in[6];
    const float* ln_offset = (const float*)d_in[7];
    const float* W_p1      = (const float*)d_in[8];
    const float* b_p1      = (const float*)d_in[9];
    const float* W_p2      = (const float*)d_in[10];
    const float* b_p2      = (const float*)d_in[11];
    const float* W_q1      = (const float*)d_in[12];  // [2048,1024]
    const float* b_q1      = (const float*)d_in[13];
    const float* W_q2      = (const float*)d_in[14];
    const float* b_q2      = (const float*)d_in[15];

    float* out    = (float*)d_out;
    float* priors = out;
    float* posts  = out + (size_t)TT * BB * 1024;
    float* pfeat  = out + (size_t)2 * TT * BB * 1024;
    float* qfeat  = pfeat + (size_t)TT * BB * 2048;

    // ---- workspace carve-out ----
    char* wsp = (char*)d_ws;
    auto carve = [&](size_t bytes) -> char* {
        char* p = wsp;
        wsp += (bytes + 255) & ~(size_t)255;
        return p;
    };
    bf16_t* WpreT = (bf16_t*)carve((size_t)1024 * KPRE_PAD * 2);
    bf16_t* WgruT = (bf16_t*)carve((size_t)3072 * 2048 * 2);
    bf16_t* Wp1T  = (bf16_t*)carve((size_t)1024 * 1024 * 2);
    bf16_t* Wp2T  = (bf16_t*)carve((size_t)1024 * 1024 * 2);
    bf16_t* Wq1T  = (bf16_t*)carve((size_t)1024 * 2048 * 2);
    bf16_t* Wq2T  = (bf16_t*)carve((size_t)1024 * 1024 * 2);
    float* state_pr = (float*)carve((size_t)BB * 2048 * 4);
    float* state_po = (float*)carve((size_t)BB * 2048 * 4);
    float* xin    = (float*)carve((size_t)BB * KPRE_PAD * 4);
    float* pre    = (float*)carve((size_t)BB * 1024 * 4);
    float* gru_in = (float*)carve((size_t)BB * 2048 * 4);
    float* parts  = (float*)carve((size_t)BB * 3072 * 4);
    float* h1     = (float*)carve((size_t)BB * 1024 * 4);
    float* q_in   = (float*)carve((size_t)BB * 2048 * 4);

    // ---- init: zero states, convert+transpose weights to bf16 ----
    {
        long n = (long)BB * 2048;
        zero_f32<<<(unsigned)((n + 255) / 256), 256, 0, stream>>>(state_pr, n);
        zero_f32<<<(unsigned)((n + 255) / 256), 256, 0, stream>>>(state_po, n);
    }
    auto conv = [&](const float* W, bf16_t* Wt, int K, int N, int Kpad) {
        long total = (long)N * Kpad;
        convert_wt<<<(unsigned)((total + 255) / 256), 256, 0, stream>>>(W, Wt, K, N, Kpad);
    };
    conv(W_pre, WpreT, 1030, 1024, KPRE_PAD);
    conv(W_gru, WgruT, 2048, 3072, 2048);
    conv(W_p1,  Wp1T,  1024, 1024, 1024);
    conv(W_p2,  Wp2T,  1024, 1024, 1024);
    conv(W_q1,  Wq1T,  2048, 1024, 2048);
    conv(W_q2,  Wq2T,  1024, 1024, 1024);

    auto gemm = [&](const float* A, int lda, const bf16_t* Wt, int K,
                    const float* bias, float* C, int N, int act) {
        dim3 grid(N / GTN, BB / GTM);
        gemm_bf16_wmma<<<grid, 256, 0, stream>>>(A, lda, Wt, K, bias, C, N, act);
    };
    const unsigned nCat = (BB * 2048 + 255) / 256;
    const unsigned nXin = (BB * KPRE_PAD + 255) / 256;

    // ================= prior scan =================
    for (int t = 0; t < TT; ++t) {
        const float* act_t = actions + (size_t)t * BB * AA;
        concat_xin<<<nXin, 256, 0, stream>>>(state_pr, act_t, xin);
        gemm(xin, KPRE_PAD, WpreT, KPRE_PAD, b_pre, pre, 1024, 1);          // x = elu(...)
        concat2<<<nCat, 256, 0, stream>>>(pre, 1024, 0, state_pr, 2048, gru_in);
        gemm(gru_in, 2048, WgruT, 2048, b_gru, parts, 3072, 0);             // GRU matmul
        gru_ln_kernel<<<BB, 256, 0, stream>>>(parts, ln_scale, ln_offset,
                                              state_pr, pfeat + (size_t)t * BB * 2048);
        gemm(state_pr, 2048, Wp1T, 1024, b_p1, h1, 1024, 1);                // elu(deter@Wp1)
        float* stats = priors + (size_t)t * BB * 1024;
        gemm(h1, 1024, Wp2T, 1024, b_p2, stats, 1024, 0);                   // stats
        sample_kernel<<<BB, 256, 0, stream>>>(stats, state_pr,
                                              pfeat + (size_t)t * BB * 2048,
                                              0x13572468u + (unsigned)t * 0x01000193u);
    }

    // ================= posterior scan =================
    // (prior-branch stats/sample inside post_scan are dead code: only deter is used)
    for (int t = 0; t < TT; ++t) {
        const float* act_t = actions + (size_t)t * BB * AA;
        const float* emb_t = embeds  + (size_t)t * BB * EE;
        concat_xin<<<nXin, 256, 0, stream>>>(state_po, act_t, xin);
        gemm(xin, KPRE_PAD, WpreT, KPRE_PAD, b_pre, pre, 1024, 1);
        concat2<<<nCat, 256, 0, stream>>>(pre, 1024, 0, state_po, 2048, gru_in);
        gemm(gru_in, 2048, WgruT, 2048, b_gru, parts, 3072, 0);
        gru_ln_kernel<<<BB, 256, 0, stream>>>(parts, ln_scale, ln_offset,
                                              state_po, qfeat + (size_t)t * BB * 2048);
        concat2<<<nCat, 256, 0, stream>>>(state_po, 2048, 0, emb_t, 1024, q_in);
        gemm(q_in, 2048, Wq1T, 2048, b_q1, h1, 1024, 1);                    // elu([deter,emb]@Wq1)
        float* stats = posts + (size_t)t * BB * 1024;
        gemm(h1, 1024, Wq2T, 1024, b_q2, stats, 1024, 0);
        sample_kernel<<<BB, 256, 0, stream>>>(stats, state_po,
                                              qfeat + (size_t)t * BB * 2048,
                                              0x24681357u + (unsigned)t * 0x01000193u);
    }
}
